// GroupedQueryAttention_73212012527885
// MI455X (gfx1250) — matrix operand order repaired and checked by reading, not compiled, not run on hardware
//
#include <hip/hip_runtime.h>
#include <hip/hip_bf16.h>

// ---------------------------------------------------------------------------
// GQA attention block for MI455X (gfx1250, wave32), bf16 WMMA everywhere.
// ---------------------------------------------------------------------------

typedef __attribute__((ext_vector_type(16))) __bf16 v16bf;
typedef __attribute__((ext_vector_type(8)))  __bf16 v8bf;
typedef __attribute__((ext_vector_type(8)))  float  v8f;

#define SEQ      2048
#define HID      2048
#define NHEADS   32
#define NKVHEADS 8
#define NGROUPS  (NHEADS / NKVHEADS)
#define HDIM     64
#define KVDIM    (NKVHEADS * HDIM)   // 512
#define QK_SCALE 0.125f              // 1/sqrt(64)

// ---------------------------------------------------------------------------
// WMMA helpers (CDNA5 16x16x32 bf16, f32 accumulate)
// ---------------------------------------------------------------------------
__device__ __forceinline__ v8f wmma_bf16(v16bf a, v16bf b, v8f c) {
  return __builtin_amdgcn_wmma_f32_16x16x32_bf16(
      /*neg_a=*/false, a, /*neg_b=*/false, b,
      /*c_mod=*/(short)0, c, /*reuse_a=*/false, /*reuse_b=*/false);
}

// A fragment, 16x32 bf16 (MxK), row-major source with leading dim ld.
// Lanes 0-15: row = row0+lane, K {kb+0..15}
// Lanes16-31: row = row0+lane-16, K {kb+16..31}
__device__ __forceinline__ v16bf load_a_frag(const __bf16* base, int ld,
                                             int row0, int kb, int lane) {
  int grp = lane >> 4;
  const __bf16* p = base + (size_t)(row0 + (lane & 15)) * ld + kb + grp * 16;
  v8bf lo = *reinterpret_cast<const v8bf*>(p);
  v8bf hi = *reinterpret_cast<const v8bf*>(p + 8);
  return __builtin_shufflevector(lo, hi, 0,1,2,3,4,5,6,7,8,9,10,11,12,13,14,15);
}

// B fragment, 32x16 bf16 (KxN). Column n of B == row (col0+n) of the stored
// row-major matrix (ld = its row length).  Lanes 0-15 hold K kb..kb+15,
// lanes 16-31 hold K kb+16..kb+31 -> one contiguous 32B load per lane.
__device__ __forceinline__ v16bf load_b_frag(const __bf16* base, int ld,
                                             int col0, int kb, int lane) {
  int grp = lane >> 4;
  const __bf16* p = base + (size_t)(col0 + (lane & 15)) * ld + kb + grp * 16;
  return *reinterpret_cast<const v16bf*>(p);
}

// 16-lane butterfly reductions (rows of a C tile live within a 16-lane group)
__device__ __forceinline__ float rowmax16(float v) {
  v = fmaxf(v, __shfl_xor(v, 1, 32));
  v = fmaxf(v, __shfl_xor(v, 2, 32));
  v = fmaxf(v, __shfl_xor(v, 4, 32));
  v = fmaxf(v, __shfl_xor(v, 8, 32));
  return v;
}
__device__ __forceinline__ float rowsum16(float v) {
  v += __shfl_xor(v, 1, 32);
  v += __shfl_xor(v, 2, 32);
  v += __shfl_xor(v, 4, 32);
  v += __shfl_xor(v, 8, 32);
  return v;
}

// ---------------------------------------------------------------------------
// f32 -> bf16 conversion
// ---------------------------------------------------------------------------
__global__ void cvt_f32_bf16_kernel(const float* __restrict__ x,
                                    __bf16* __restrict__ y, long n) {
  long i = (long)blockIdx.x * blockDim.x + threadIdx.x;
  long stride = (long)gridDim.x * blockDim.x;
  for (; i < n; i += stride) y[i] = (__bf16)x[i];
}

// ---------------------------------------------------------------------------
// GEMM: C[M,N] (f32) = A[M,K] (bf16, ld=K) x W[N,K]^T (bf16, row-major)
// 256 threads = 8 waves (4M x 2N), wave tile 32x64 (2x4 WMMA tiles),
// block tile 128x128.  M%128==0, N%128==0, K%32==0 for all calls here.
// Per k-step: 8 WMMAs on 2 A-frags + 4 B-frags (12 b128 loads / lane).
// ---------------------------------------------------------------------------
__global__ __launch_bounds__(256)
void gemm_bf16_kernel(const __bf16* __restrict__ A, const __bf16* __restrict__ W,
                      float* __restrict__ C, int M, int N, int K) {
  int lane = threadIdx.x & 31;
  int wv   = threadIdx.x >> 5;          // 0..7
  int wm   = wv & 3;                    // 4 waves over M
  int wn   = wv >> 2;                   // 2 waves over N
  int m0 = blockIdx.x * 128 + wm * 32;
  int n0 = blockIdx.y * 128 + wn * 64;

  v8f acc[2][4] = {};
  for (int kb = 0; kb < K; kb += 32) {
    v16bf a0 = load_a_frag(A, K, m0,      kb, lane);
    v16bf a1 = load_a_frag(A, K, m0 + 16, kb, lane);
    v16bf b0 = load_b_frag(W, K, n0,      kb, lane);
    v16bf b1 = load_b_frag(W, K, n0 + 16, kb, lane);
    v16bf b2 = load_b_frag(W, K, n0 + 32, kb, lane);
    v16bf b3 = load_b_frag(W, K, n0 + 48, kb, lane);
    acc[0][0] = wmma_bf16(a0, b0, acc[0][0]);
    acc[0][1] = wmma_bf16(a0, b1, acc[0][1]);
    acc[0][2] = wmma_bf16(a0, b2, acc[0][2]);
    acc[0][3] = wmma_bf16(a0, b3, acc[0][3]);
    acc[1][0] = wmma_bf16(a1, b0, acc[1][0]);
    acc[1][1] = wmma_bf16(a1, b1, acc[1][1]);
    acc[1][2] = wmma_bf16(a1, b2, acc[1][2]);
    acc[1][3] = wmma_bf16(a1, b3, acc[1][3]);
  }
  int col = lane & 15;
  int rb  = (lane >> 4) * 8;
#pragma unroll
  for (int i = 0; i < 2; ++i) {
#pragma unroll
    for (int j = 0; j < 4; ++j) {
      size_t base = (size_t)(m0 + i * 16 + rb) * N + n0 + j * 16 + col;
#pragma unroll
      for (int r = 0; r < 8; ++r) C[base + (size_t)r * N] = acc[i][j][r];
    }
  }
}

// ---------------------------------------------------------------------------
// RoPE cos/sin table: tab[s*32+j] = {cos, sin} of s * theta^(-j/32)
// ---------------------------------------------------------------------------
__global__ void rope_table_kernel(float* __restrict__ tab) {
  int i = blockIdx.x * blockDim.x + threadIdx.x;     // SEQ*32 entries
  if (i >= SEQ * 32) return;
  int s = i >> 5, j = i & 31;
  float inv = __powf(10000.0f, -(float)j * (1.0f / 32.0f));
  float ang = (float)s * inv;
  tab[2 * i]     = __cosf(ang);
  tab[2 * i + 1] = __sinf(ang);
}

// ---------------------------------------------------------------------------
// RoPE (f32 in -> bf16 out), layout [B*S, ncols] where col = head*64 + d
// ---------------------------------------------------------------------------
__global__ void rope_bf16_kernel(const float* __restrict__ X,
                                 const float* __restrict__ tab,
                                 __bf16* __restrict__ Y, int ncols, long n) {
  long idx = (long)blockIdx.x * blockDim.x + threadIdx.x;
  if (idx >= n) return;
  int c = (int)(idx % ncols);
  long m = idx / ncols;
  int s = (int)(m % SEQ);
  int d = c & (HDIM - 1);
  int j = d & 31;
  float cs = tab[2 * (s * 32 + j)];
  float sn = tab[2 * (s * 32 + j) + 1];
  float x = X[idx];
  float other = (d < 32) ? -X[idx + 32] : X[idx - 32];
  Y[idx] = (__bf16)(x * cs + other * sn);
}

// ---------------------------------------------------------------------------
// V transpose: V[B*S, 512] f32 -> Vt[B, 8, 64, S] bf16  (PV B-frags contiguous)
// ---------------------------------------------------------------------------
__global__ void vtrans_kernel(const float* __restrict__ V,
                              __bf16* __restrict__ Vt, long n) {
  long idx = (long)blockIdx.x * blockDim.x + threadIdx.x;
  if (idx >= n) return;
  int c = (int)(idx % KVDIM);
  long m = idx / KVDIM;
  int b = (int)(m / SEQ), s = (int)(m % SEQ);
  int kv = c >> 6, d = c & 63;
  Vt[(((size_t)b * NKVHEADS + kv) * HDIM + d) * SEQ + s] = (__bf16)V[idx];
}

// ---------------------------------------------------------------------------
// Flash attention, causal, GQA (4 q-heads share a kv-head).
// Block = 128 threads = 4 waves; each wave owns 16 query rows.
// k-tiles of 32 columns; per tile: 4 WMMA for scores, 4 WMMA for PV.
// ---------------------------------------------------------------------------
__global__ __launch_bounds__(128)
void attn_kernel(const __bf16* __restrict__ Qb, const __bf16* __restrict__ Kb,
                 const __bf16* __restrict__ Vt, __bf16* __restrict__ Ab) {
  __shared__ __align__(16) __bf16 pbuf[4][16 * 32];   // per-wave P staging

  int lane = threadIdx.x & 31;
  int wv   = threadIdx.x >> 5;          // 0..3
  const int nqt = SEQ / 64;
  int qt = blockIdx.x % nqt;
  int bh = blockIdx.x / nqt;
  int h  = bh % NHEADS;
  int b  = bh / NHEADS;
  int kv = h / NGROUPS;
  int q0 = qt * 64 + wv * 16;           // first q row of this wave

  const __bf16* Qbase = Qb + (size_t)b * SEQ * HID + (size_t)h * HDIM;
  const __bf16* Kbase = Kb + (size_t)b * SEQ * KVDIM + (size_t)kv * HDIM;
  const __bf16* Vbase = Vt + ((size_t)b * NKVHEADS + kv) * HDIM * SEQ;

  // Q fragments (16 rows x 64 d), two K-chunks of 32
  v16bf qa0 = load_a_frag(Qbase, HID, q0, 0,  lane);
  v16bf qa1 = load_a_frag(Qbase, HID, q0, 32, lane);

  v8f o0 = {}, o1 = {}, o2 = {}, o3 = {};
  float m_i[8], l_i[8];
#pragma unroll
  for (int r = 0; r < 8; ++r) { m_i[r] = -__builtin_inff(); l_i[r] = 0.0f; }

  int colBase = lane & 15;
  int rloc    = (lane >> 4) * 8;        // local row base (0 or 8)
  int rowA    = q0 + rloc;              // absolute q row of element r = rowA+r
  __bf16* pl  = pbuf[wv];

  for (int k0 = 0; k0 <= q0 + 15; k0 += 32) {
    // ---- scores S = (Q K^T) * scale, two 16-col tiles ----
    v8f s0 = {}, s1 = {};
    s0 = wmma_bf16(qa0, load_b_frag(Kbase, KVDIM, k0,      0,  lane), s0);
    s0 = wmma_bf16(qa1, load_b_frag(Kbase, KVDIM, k0,      32, lane), s0);
    s1 = wmma_bf16(qa0, load_b_frag(Kbase, KVDIM, k0 + 16, 0,  lane), s1);
    s1 = wmma_bf16(qa1, load_b_frag(Kbase, KVDIM, k0 + 16, 32, lane), s1);

    // ---- causal mask + online softmax ----
    float fac[8], mnew[8];
#pragma unroll
    for (int r = 0; r < 8; ++r) {
      int row = rowA + r;
      float a = s0[r] * QK_SCALE;
      float c = s1[r] * QK_SCALE;
      if (k0 + colBase > row)      a = -__builtin_inff();
      if (k0 + 16 + colBase > row) c = -__builtin_inff();
      s0[r] = a; s1[r] = c;
      float mx = rowmax16(fmaxf(a, c));
      mnew[r] = fmaxf(m_i[r], mx);
      fac[r] = __expf(m_i[r] - mnew[r]);
      m_i[r] = mnew[r];
    }
#pragma unroll
    for (int r = 0; r < 8; ++r) {
      float p0 = __expf(s0[r] - mnew[r]);
      float p1 = __expf(s1[r] - mnew[r]);
      l_i[r] = l_i[r] * fac[r] + rowsum16(p0 + p1);
      o0[r] *= fac[r]; o1[r] *= fac[r]; o2[r] *= fac[r]; o3[r] *= fac[r];
      // stage P into LDS (C-layout -> row-major 16x32)
      pl[(rloc + r) * 32 + colBase]      = (__bf16)p0;
      pl[(rloc + r) * 32 + 16 + colBase] = (__bf16)p1;
    }
    asm volatile("s_wait_dscnt 0" ::: "memory");

    // reload P as A-fragment (16x32), then O += P @ V
    v16bf pa = load_a_frag(pl, 32, 0, 0, lane);
    o0 = wmma_bf16(pa, load_b_frag(Vbase, SEQ, 0,  k0, lane), o0);
    o1 = wmma_bf16(pa, load_b_frag(Vbase, SEQ, 16, k0, lane), o1);
    o2 = wmma_bf16(pa, load_b_frag(Vbase, SEQ, 32, k0, lane), o2);
    o3 = wmma_bf16(pa, load_b_frag(Vbase, SEQ, 48, k0, lane), o3);
  }

  // ---- normalize and store (bf16, layout [B*S, HID] at col h*64) ----
#pragma unroll
  for (int r = 0; r < 8; ++r) {
    float inv_l = 1.0f / l_i[r];
    size_t rowOff = (size_t)(b * SEQ + rowA + r) * HID + (size_t)h * HDIM;
    Ab[rowOff + 0  + colBase] = (__bf16)(o0[r] * inv_l);
    Ab[rowOff + 16 + colBase] = (__bf16)(o1[r] * inv_l);
    Ab[rowOff + 32 + colBase] = (__bf16)(o2[r] * inv_l);
    Ab[rowOff + 48 + colBase] = (__bf16)(o3[r] * inv_l);
  }
}

// ---------------------------------------------------------------------------
// Host-side orchestration
// ---------------------------------------------------------------------------
static inline size_t ws_take(size_t& off, size_t bytes) {
  size_t p = off;
  off = (off + bytes + 255) & ~(size_t)255;
  return p;
}

extern "C" void kernel_launch(void* const* d_in, const int* in_sizes, int n_in,
                              void* d_out, int out_size, void* d_ws, size_t ws_size,
                              hipStream_t stream) {
  const float* hidden = (const float*)d_in[0];
  const float* Wq = (const float*)d_in[1];
  const float* Wk = (const float*)d_in[2];
  const float* Wv = (const float*)d_in[3];
  const float* Wo = (const float*)d_in[4];

  const int M = in_sizes[0] / HID;      // B*S = 4096
  const int B = M / SEQ;                // 2

  char* ws = (char*)d_ws;
  size_t off = 0;
  __bf16* hb  = (__bf16*)(ws + ws_take(off, (size_t)M * HID * 2));
  __bf16* wqb = (__bf16*)(ws + ws_take(off, (size_t)HID * HID * 2));
  __bf16* wkb = (__bf16*)(ws + ws_take(off, (size_t)KVDIM * HID * 2));
  __bf16* wvb = (__bf16*)(ws + ws_take(off, (size_t)KVDIM * HID * 2));
  __bf16* wob = (__bf16*)(ws + ws_take(off, (size_t)HID * HID * 2));
  float*  qf  = (float*) (ws + ws_take(off, (size_t)M * HID * 4));
  float*  kf  = (float*) (ws + ws_take(off, (size_t)M * KVDIM * 4));
  float*  vf  = (float*) (ws + ws_take(off, (size_t)M * KVDIM * 4));
  __bf16* qb  = (__bf16*)(ws + ws_take(off, (size_t)M * HID * 2));
  __bf16* kb  = (__bf16*)(ws + ws_take(off, (size_t)M * KVDIM * 2));
  __bf16* vtb = (__bf16*)(ws + ws_take(off, (size_t)M * KVDIM * 2));
  __bf16* ab  = (__bf16*)(ws + ws_take(off, (size_t)M * HID * 2));
  float*  tab = (float*) (ws + ws_take(off, (size_t)SEQ * 32 * 2 * 4));

  auto cvt = [&](const float* src, __bf16* dst, long n) {
    int blocks = (int)((n + 255) / 256);
    cvt_f32_bf16_kernel<<<blocks, 256, 0, stream>>>(src, dst, n);
  };
  cvt(hidden, hb,  (long)M * HID);
  cvt(Wq,     wqb, (long)HID * HID);
  cvt(Wk,     wkb, (long)KVDIM * HID);
  cvt(Wv,     wvb, (long)KVDIM * HID);
  cvt(Wo,     wob, (long)HID * HID);

  // RoPE cos/sin table (tiny; overlaps with conversions)
  rope_table_kernel<<<(SEQ * 32 + 255) / 256, 256, 0, stream>>>(tab);

  // Q/K/V projections (WMMA GEMMs)
  gemm_bf16_kernel<<<dim3(M / 128, HID / 128),   256, 0, stream>>>(hb, wqb, qf, M, HID,   HID);
  gemm_bf16_kernel<<<dim3(M / 128, KVDIM / 128), 256, 0, stream>>>(hb, wkb, kf, M, KVDIM, HID);
  gemm_bf16_kernel<<<dim3(M / 128, KVDIM / 128), 256, 0, stream>>>(hb, wvb, vf, M, KVDIM, HID);

  // RoPE (Q, K) -> bf16 ; V -> transposed bf16
  {
    long nq = (long)M * HID;
    rope_bf16_kernel<<<(int)((nq + 255) / 256), 256, 0, stream>>>(qf, tab, qb, HID, nq);
    long nk = (long)M * KVDIM;
    rope_bf16_kernel<<<(int)((nk + 255) / 256), 256, 0, stream>>>(kf, tab, kb, KVDIM, nk);
    vtrans_kernel<<<(int)((nk + 255) / 256), 256, 0, stream>>>(vf, vtb, nk);
  }

  // Fused causal flash attention
  {
    int blocks = B * NHEADS * (SEQ / 64);
    attn_kernel<<<blocks, 128, 0, stream>>>(qb, kb, vtb, ab);
  }

  // Output projection into f32 d_out
  gemm_bf16_kernel<<<dim3(M / 128, HID / 128), 256, 0, stream>>>(ab, wob, (float*)d_out,
                                                                 M, HID, HID);
}
